// MultiHeadAttention_20126216749240
// MI455X (gfx1250) — matrix operand order repaired and checked by reading, not compiled, not run on hardware
//
#include <hip/hip_runtime.h>
#include <hip/hip_bf16.h>

typedef __attribute__((ext_vector_type(16))) _Float16 v16h;
typedef __attribute__((ext_vector_type(8)))  _Float16 v8h;
typedef __attribute__((ext_vector_type(8)))  float    v8f;

#define BATCH 8
#define SEQ   2048
#define EMB   2048
#define HEADS 32
#define HD    64
#define MTOT  (BATCH * SEQ)          // 16384 rows of the output GEMM
#define SC    16                     // sequence chunks for key column-sum
#define SCL   (SEQ / SC)             // 128

// ---------------------------------------------------------------------------
// Stage 0: f32 -> f16 conversion (Wo), one element per thread
// ---------------------------------------------------------------------------
__global__ void mha_f32_to_f16(const float* __restrict__ in,
                               _Float16* __restrict__ out, int n) {
    int i = blockIdx.x * blockDim.x + threadIdx.x;
    if (i < n) out[i] = (_Float16)in[i];
}

// ---------------------------------------------------------------------------
// Stage 1a: partial column sums of keys over sequence chunks (deterministic)
// part[sc][b][e] = sum_{s in chunk sc} keys[b][s][e]
// grid: (EMB/256, BATCH, SC), block 256
// ---------------------------------------------------------------------------
__global__ void mha_keys_colsum_partial(const float* __restrict__ keys,
                                        float* __restrict__ part) {
    int e  = blockIdx.x * 256 + threadIdx.x;
    int b  = blockIdx.y;
    int sc = blockIdx.z;
    const float* p = keys + ((size_t)b * SEQ + (size_t)sc * SCL) * EMB + e;
    float s = 0.0f;
#pragma unroll 8
    for (int i = 0; i < SCL; ++i) s += p[(size_t)i * EMB];
    part[((size_t)sc * BATCH + b) * EMB + e] = s;
}

// ---------------------------------------------------------------------------
// Stage 1b: reduce the SC partials -> raw[b][e]   (16384 elements)
// ---------------------------------------------------------------------------
__global__ void mha_keys_colsum_final(const float* __restrict__ part,
                                      float* __restrict__ raw) {
    int i = blockIdx.x * 256 + threadIdx.x;   // 0 .. BATCH*EMB-1
    float s = 0.0f;
#pragma unroll
    for (int c = 0; c < SC; ++c) s += part[(size_t)c * (BATCH * EMB) + i];
    raw[i] = s;
}

// ---------------------------------------------------------------------------
// Stage 1c: ksum_lin[b][h][d] = Wk[d][:] . raw[b][h*64 .. +63] + S * bk[d]
// ---------------------------------------------------------------------------
__global__ void mha_project_ksum(const float* __restrict__ raw,
                                 const float* __restrict__ Wk,
                                 const float* __restrict__ bk,
                                 float* __restrict__ kl) {
    int i = blockIdx.x * 256 + threadIdx.x;   // 0 .. BATCH*EMB-1
    int b = i >> 11;
    int e = i & (EMB - 1);
    int h = e >> 6;
    int d = e & (HD - 1);
    const float* rp = raw + (size_t)b * EMB + h * HD;
    const float* w  = Wk + d * HD;
    float s = (float)SEQ * bk[d];
#pragma unroll 16
    for (int j = 0; j < HD; ++j) s = fmaf(w[j], rp[j], s);
    kl[i] = s;
}

// ---------------------------------------------------------------------------
// Stage 2: per (token, head): project q and v (64x64 matvec), dot with key
// sum, softmax over head_dim (64), multiply by v -> att (f16).
// Block = 256 threads = 4 groups of 64 (one group per (token,head)).
// ---------------------------------------------------------------------------
__global__ __launch_bounds__(256) void mha_attention_stage(
    const float* __restrict__ queries, const float* __restrict__ values,
    const float* __restrict__ Wq, const float* __restrict__ bq,
    const float* __restrict__ Wv, const float* __restrict__ bv,
    const float* __restrict__ kl, _Float16* __restrict__ att) {
    __shared__ float shq[4][HD];
    __shared__ float shv[4][HD];
    __shared__ float red[256];

    const int g = threadIdx.x >> 6;           // group 0..3
    const int t = threadIdx.x & 63;           // head-dim index
    const int gid   = blockIdx.x * 4 + g;     // 0 .. B*S*HEADS-1
    const int head  = gid & (HEADS - 1);
    const int token = gid >> 5;               // 0 .. 16383
    const int b     = token >> 11;
    const size_t base = (size_t)token * EMB + head * HD;

    shq[g][t] = queries[base + t];
    shv[g][t] = values[base + t];
    __syncthreads();

    float qp = bq[t];
    float vp = bv[t];
    const float* wq = Wq + t * HD;
    const float* wv = Wv + t * HD;
#pragma unroll 16
    for (int j = 0; j < HD; ++j) {
        qp = fmaf(wq[j], shq[g][j], qp);
        vp = fmaf(wv[j], shv[g][j], vp);
    }

    const float ks  = kl[(size_t)b * EMB + head * HD + t];
    const float dot = qp * ks * 0.02209708691207961f;   // 1/sqrt(2048)

    // softmax over the 64 lanes of this group (max, then sum) via LDS
    red[threadIdx.x] = dot;
    __syncthreads();
    for (int off = 32; off > 0; off >>= 1) {
        if (t < off) red[g * 64 + t] = fmaxf(red[g * 64 + t], red[g * 64 + t + off]);
        __syncthreads();
    }
    const float mx = red[g * 64];
    __syncthreads();
    const float ev = __expf(dot - mx);
    red[threadIdx.x] = ev;
    __syncthreads();
    for (int off = 32; off > 0; off >>= 1) {
        if (t < off) red[g * 64 + t] += red[g * 64 + t + off];
        __syncthreads();
    }
    const float sum = red[g * 64];

    att[base + t] = (_Float16)((ev / sum) * vp);
}

// ---------------------------------------------------------------------------
// Stage 3: out(16384x2048) = att(16384x2048) @ Wo^T + bo via WMMA f16.
// Block tile 128M x 128N, 8 waves; wave tile 32M x 64N = 2x4 WMMA C tiles.
// Ping-pong double buffering (K unrolled by 2, 64 per iteration): each half
// computes from one fragment set and prefetches 64 ahead into the same set,
// so there are no inter-buffer register copies (no v_mov rotation, no
// WMMA->VALU hazard NOPs).
//
// A fragment (16x32 f16): lane (m=l&15, hi=l>>4) holds K {16hi..16hi+15}
// -> two contiguous b128 loads from row-major att.
// B fragment (32x16 f16): lane (n=l&15, hi=l>>4) holds K {16hi..16hi+15},
// contiguous along a Wo row (B = Wo^T) -> one 32B load.
// C/D: row = vgpr + 8*hi, col = lane&15.
// ---------------------------------------------------------------------------
__device__ __forceinline__ void mha_load_frags(
    const _Float16* __restrict__ aBase0,   // row (mi=0) pointer at lane K offset
    const _Float16* __restrict__ aBase1,   // row (mi=1) pointer at lane K offset
    const _Float16* __restrict__ bBase0,   // Wo rows for ni=0..3 at lane K offset
    int k0, v16h a[2], v16h b[4]) {
#pragma unroll
    for (int mi = 0; mi < 2; ++mi) {
        const _Float16* ap = (mi == 0 ? aBase0 : aBase1) + k0;
        v8h alo = *(const v8h*)ap;
        v8h ahi = *(const v8h*)(ap + 8);
#pragma unroll
        for (int tt = 0; tt < 8; ++tt) {
            a[mi][tt]     = alo[tt];
            a[mi][tt + 8] = ahi[tt];
        }
    }
#pragma unroll
    for (int ni = 0; ni < 4; ++ni)
        b[ni] = *(const v16h*)(bBase0 + (size_t)ni * 16 * EMB + k0);
}

__device__ __forceinline__ void mha_wmma8(v8f acc[2][4], const v16h a[2],
                                          const v16h b[4]) {
#pragma unroll
    for (int mi = 0; mi < 2; ++mi)
#pragma unroll
        for (int ni = 0; ni < 4; ++ni)
            acc[mi][ni] = __builtin_amdgcn_wmma_f32_16x16x32_f16(
                false, a[mi], false, b[ni], (short)0, acc[mi][ni],
                false, false);
}

__global__ __launch_bounds__(256) void mha_out_gemm(
    const _Float16* __restrict__ A,      // MTOT x EMB (att)
    const _Float16* __restrict__ Bw,     // EMB x EMB row-major (Wo in f16)
    const float* __restrict__ bias,      // bo
    float* __restrict__ C) {
    const int bm   = blockIdx.x * 128;
    const int bn   = blockIdx.y * 128;
    const int wave = threadIdx.x >> 5;       // 0..7
    const int lane = threadIdx.x & 31;
    const int lo   = lane & 15;
    const int hi   = lane >> 4;
    const int wm   = (wave & 3) * 32;        // 4 waves along M
    const int wn   = (wave >> 2) * 64;       // 2 waves along N

    // Per-lane base pointers (K offset folded in once).
    const _Float16* aBase0 = A  + (size_t)(bm + wm +  0 + lo) * EMB + 16 * hi;
    const _Float16* aBase1 = A  + (size_t)(bm + wm + 16 + lo) * EMB + 16 * hi;
    const _Float16* bBase0 = Bw + (size_t)(bn + wn + lo) * EMB + 16 * hi;

    v8f acc[2][4] = {};
    v16h a0[2], b0[4], a1[2], b1[4];
    mha_load_frags(aBase0, aBase1, bBase0, 0,  a0, b0);
    mha_load_frags(aBase0, aBase1, bBase0, 32, a1, b1);

    for (int k0 = 0; k0 < EMB - 64; k0 += 64) {
        mha_wmma8(acc, a0, b0);
        mha_load_frags(aBase0, aBase1, bBase0, k0 + 64, a0, b0);  // prefetch
        mha_wmma8(acc, a1, b1);
        mha_load_frags(aBase0, aBase1, bBase0, k0 + 96, a1, b1);  // prefetch
    }
    mha_wmma8(acc, a0, b0);   // K = 1984
    mha_wmma8(acc, a1, b1);   // K = 2016

#pragma unroll
    for (int mi = 0; mi < 2; ++mi)
#pragma unroll
        for (int ni = 0; ni < 4; ++ni) {
            const int col = bn + wn + ni * 16 + lo;
            const float bv = bias[col];
#pragma unroll
            for (int j = 0; j < 8; ++j) {
                const int row = bm + wm + mi * 16 + j + 8 * hi;
                C[(size_t)row * EMB + col] = acc[mi][ni][j] + bv;
            }
        }
}

// ---------------------------------------------------------------------------
// Launch
// ---------------------------------------------------------------------------
extern "C" void kernel_launch(void* const* d_in, const int* in_sizes, int n_in,
                              void* d_out, int out_size, void* d_ws, size_t ws_size,
                              hipStream_t stream) {
    const float* values  = (const float*)d_in[0];
    const float* keys    = (const float*)d_in[1];
    const float* queries = (const float*)d_in[2];
    // d_in[3] = mask : faithfully ignored (reference discards masked_fill)
    const float* Wk = (const float*)d_in[4];
    const float* bk = (const float*)d_in[5];
    const float* Wv = (const float*)d_in[6];
    const float* bv = (const float*)d_in[7];
    const float* Wq = (const float*)d_in[8];
    const float* bq = (const float*)d_in[9];
    const float* Wo = (const float*)d_in[10];
    const float* bo = (const float*)d_in[11];
    float* out = (float*)d_out;

    char* ws = (char*)d_ws;
    _Float16* Wo_h = (_Float16*)ws;                          //  8 MB
    _Float16* att  = (_Float16*)(ws + (size_t)8388608);      // 64 MB
    float*    part = (float*)(ws + (size_t)75497472);        //  2 MB
    float*    raw  = (float*)(ws + (size_t)77594624);        // 64 KB
    float*    kl   = (float*)(ws + (size_t)77660160);        // 64 KB

    // Wo -> f16 (4M elements)
    mha_f32_to_f16<<<dim3(4194304 / 256), dim3(256), 0, stream>>>(
        Wo, Wo_h, 4194304);

    // key column-sum (two-phase, deterministic) + tiny projection
    mha_keys_colsum_partial<<<dim3(EMB / 256, BATCH, SC), dim3(256), 0, stream>>>(
        keys, part);
    mha_keys_colsum_final<<<dim3((BATCH * EMB) / 256), dim3(256), 0, stream>>>(
        part, raw);
    mha_project_ksum<<<dim3((BATCH * EMB) / 256), dim3(256), 0, stream>>>(
        raw, Wk, bk, kl);

    // fused q/v projection + head-dim softmax -> att (f16)
    mha_attention_stage<<<dim3((BATCH * SEQ * HEADS) / 4), dim3(256), 0, stream>>>(
        queries, values, Wq, bq, Wv, bv, kl, att);

    // output projection GEMM with WMMA
    mha_out_gemm<<<dim3(MTOT / 128, EMB / 128), dim3(256), 0, stream>>>(
        att, Wo_h, bo, out);
}
